// GSPN_22101901705273
// MI455X (gfx1250) — matrix operand order repaired and checked by reading, not compiled, not run on hardware
//
#include <hip/hip_runtime.h>

// ---------------------------------------------------------------------------
// GSPN block for MI455X (gfx1250, wave32).
// - All 1x1-conv GEMMs + gate low-rank up-projections: v_wmma_f32_16x16x32_bf16
//   (bf16 in, fp32 accumulate). The 302MB gate tensor is never materialized:
//   gates are recomputed per scan column with 6 WMMA tiles.
// - The sequential tridiagonal scan stays fp32 VALU; its strided directional
//   X reads are software-pipelined with GLOBAL_LOAD_ASYNC_TO_LDS_B32 (per-lane
//   addresses, ASYNCcnt) double-buffered in LDS, drained via s_wait_asynccnt.
// ---------------------------------------------------------------------------

typedef __attribute__((ext_vector_type(16))) __bf16 v16bf;
typedef __attribute__((ext_vector_type(8)))  float  v8f;

#define CI   192      // D_INNER
#define CIN  96       // D_MODEL
#define S16  16       // D_STATE
#define HW   16384    // 128*128
#define HH   128
#define WW   128

__device__ __forceinline__ v8f wmma_bf16(v16bf a, v16bf b, v8f c) {
  return __builtin_amdgcn_wmma_f32_16x16x32_bf16(false, a, false, b, (short)0, c,
                                                 false, false);
}

// A-matrix (M x K tile, 16x32, bf16) from row-major f32 W[m*ld + k].
// Contiguous K order (matches B loaders): lanes 0-15 row M=lane, half h -> K=h;
// lanes 16-31 row M=lane-16, half h -> K=16+h.
__device__ __forceinline__ v16bf wmma_load_a(const float* __restrict__ W, int ld,
                                             int m0, int k0, int ktot) {
  unsigned lane = threadIdx.x & 31u;
  int m = m0 + (int)(lane & 15u);
  v16bf a;
#pragma unroll
  for (int h = 0; h < 16; ++h) {
    int k = k0 + h + ((lane < 16u) ? 0 : 16);
    float v = (k < ktot) ? W[m * ld + k] : 0.0f;
    a[h] = (__bf16)v;
  }
  return a;
}

// B-matrix (K x N tile, 32x16, bf16) from row-major f32 X[k*ld + n].
// lanes 0-15: N=n0+lane, half h -> K=h; lanes 16-31: N=n0+lane-16, K=16+h.
__device__ __forceinline__ v16bf wmma_load_b(const float* __restrict__ X, int ld,
                                             int k0, int n0, int ktot) {
  unsigned lane = threadIdx.x & 31u;
  int n = n0 + (int)(lane & 15u);
  v16bf b;
#pragma unroll
  for (int h = 0; h < 16; ++h) {
    int k = k0 + h + ((lane < 16u) ? 0 : 16);
    float v = (k < ktot) ? X[k * ld + n] : 0.0f;
    b[h] = (__bf16)v;
  }
  return b;
}

__device__ __forceinline__ v16bf wmma_load_b_scaled(const float* __restrict__ X, int ld,
                                                    int k0, int n0, int ktot,
                                                    const float* __restrict__ scale) {
  unsigned lane = threadIdx.x & 31u;
  int n = n0 + (int)(lane & 15u);
  v16bf b;
#pragma unroll
  for (int h = 0; h < 16; ++h) {
    int k = k0 + h + ((lane < 16u) ? 0 : 16);
    float v = (k < ktot) ? X[k * ld + n] * scale[k] : 0.0f;
    b[h] = (__bf16)v;
  }
  return b;
}

// C/D (16x16 f32): lanes 0-15 N=n0+lane, VGPR r -> M=m0+r; lanes 16-31 -> M=m0+8+r.
__device__ __forceinline__ void wmma_store(float* __restrict__ Y, int ld,
                                           int m0, int n0, v8f c) {
  unsigned lane = threadIdx.x & 31u;
  int n  = n0 + (int)(lane & 15u);
  int mb = m0 + ((lane < 16u) ? 0 : 8);
#pragma unroll
  for (int r = 0; r < 8; ++r) Y[(mb + r) * ld + n] = c[r];
}

__device__ __forceinline__ void wmma_store_bias(float* __restrict__ Y, int ld,
                                                int m0, int n0, v8f c,
                                                const float* __restrict__ bias) {
  unsigned lane = threadIdx.x & 31u;
  int n  = n0 + (int)(lane & 15u);
  int mb = m0 + ((lane < 16u) ? 0 : 8);
#pragma unroll
  for (int r = 0; r < 8; ++r) Y[(mb + r) * ld + n] = c[r] + bias[mb + r];
}

__device__ __forceinline__ float sigmoidf(float x) {
  return 1.0f / (1.0f + __expf(-x));
}

// ---------------------------------------------------------------------------
// Kernel A: x1 = w_in(192x96) @ x[b](96x16384).  One 16x16 tile per wave.
__global__ void k_inproj(const float* __restrict__ x, const float* __restrict__ w_in,
                         float* __restrict__ x1) {
  int wave = blockIdx.x * (blockDim.x >> 5) + (threadIdx.x >> 5);
  int nt = wave & 1023;
  int rem = wave >> 10;          // 0..23
  int mt = rem % 12;
  int b  = rem / 12;
  const float* Xb = x + b * CIN * HW;
  float* Yb = x1 + b * CI * HW;
  v8f c = {};
#pragma unroll
  for (int k0 = 0; k0 < CIN; k0 += 32) {
    v16bf a  = wmma_load_a(w_in, CIN, mt * 16, k0, CIN);
    v16bf bb = wmma_load_b(Xb, HW, k0, nt * 16, CIN);
    c = wmma_bf16(a, bb, c);
  }
  wmma_store(Yb, HW, mt * 16, nt * 16, c);
}

// ---------------------------------------------------------------------------
// Kernel B: depthwise 3x3 (SAME, zero pad) + bias, fp32.
__global__ void k_dwconv(const float* __restrict__ x1, const float* __restrict__ w_dw,
                         const float* __restrict__ b_dw, float* __restrict__ x1d) {
  int idx = blockIdx.x * 256 + threadIdx.x;      // B*CI*HW threads
  int p = idx & (HW - 1);
  int c = (idx >> 14) % CI;
  int i = p >> 7, j = p & 127;
  const float* W = w_dw + c * 9;
  const float* src = x1 + (idx - p);
  float acc = b_dw[c];
#pragma unroll
  for (int di = -1; di <= 1; ++di)
#pragma unroll
    for (int dj = -1; dj <= 1; ++dj) {
      int ii = i + di, jj = j + dj;
      if (ii >= 0 && ii < HH && jj >= 0 && jj < WW)
        acc += W[(di + 1) * 3 + (dj + 1)] * src[(ii << 7) + jj];
    }
  x1d[idx] = acc;
}

// ---------------------------------------------------------------------------
// Kernel C: xp = w_xdown(16x192) @ x1d[b](192x16384).
__global__ void k_xdown(const float* __restrict__ x1d, const float* __restrict__ w_xdown,
                        float* __restrict__ xp) {
  int wave = blockIdx.x * (blockDim.x >> 5) + (threadIdx.x >> 5);
  int nt = wave & 1023;
  int b  = wave >> 10;
  const float* Xb = x1d + b * CI * HW;
  float* Yb = xp + b * S16 * HW;
  v8f c = {};
#pragma unroll
  for (int k0 = 0; k0 < CI; k0 += 32) {
    v16bf a  = wmma_load_a(w_xdown, CI, 0, k0, CI);
    v16bf bb = wmma_load_b(Xb, HW, k0, nt * 16, CI);
    c = wmma_bf16(a, bb, c);
  }
  wmma_store(Yb, HW, 0, nt * 16, c);
}

// ---------------------------------------------------------------------------
// B tile for the gate GEMM: xp column t, N = rows i, K = 16 states (zero-pad to 32).
__device__ __forceinline__ v16bf load_b_xpcol(const float* __restrict__ xpb,
                                              int i0, int t) {
  unsigned lane = threadIdx.x & 31u;
  int i = i0 + (int)(lane & 15u);
  v16bf b;
#pragma unroll
  for (int h = 0; h < 16; ++h) {
    int s = h + ((lane < 16u) ? 0 : 16);
    float v = (s < S16) ? xpb[(s << 14) + (i << 7) + t] : 0.0f;
    b[h] = (__bf16)v;
  }
  return b;
}

__device__ __forceinline__ void store_tile_lds_bf16(__bf16* gp /*[16][HH]*/, int i0, v8f c) {
  unsigned lane = threadIdx.x & 31u;
  int i  = i0 + (int)(lane & 15u);
  int mb = (lane < 16u) ? 0 : 8;
#pragma unroll
  for (int r = 0; r < 8; ++r) gp[(mb + r) * HH + i] = (__bf16)c[r];
}

__device__ __forceinline__ int xoff_dir(int n, int i, int t) {
  if (n == 0) return (i << 7) + t;               // x1
  if (n == 1) return (t << 7) + i;               // transpose
  if (n == 2) return (i << 7) + (WW - 1 - t);    // W-flip
  return ((WW - 1 - t) << 7) + i;                // transpose + flip
}

// ---------------------------------------------------------------------------
// Kernel D: fused gated tridiagonal scan for direction n.
// Block = (batch b, 16-channel tile ct), 256 threads (8 waves).
// Per column t:
//  - async prefetch of column t+1's directional X gather into LDS (double buffer)
//  - 6 WMMA tiles compute the 96 gate rows (gl,gm,gr,l,u,d) from xp[:, :, t]
//  - fp32 tridiagonal recurrence over H with LDS-carried state.
// Dynamic LDS layout (48 KB): g(bf16) 24KB | h(f32) 8KB | xcol[2](f32) 16KB.
__global__ void k_scan(const float* __restrict__ x1d, const float* __restrict__ xp,
                       const float* __restrict__ w_wup, const float* __restrict__ w_lup,
                       const float* __restrict__ w_uup, const float* __restrict__ w_ddown,
                       const float* __restrict__ w_m, float* __restrict__ y,
                       int n, int accumulate) {
  extern __shared__ char smem[];
  __bf16* g    = (__bf16*)smem;                         // [6][16][HH] bf16
  float*  h    = (float*)(smem + 6 * 16 * HH * 2);      // [16][HH]
  float*  xcol = (float*)(smem + 6 * 16 * HH * 2 + 16 * HH * 4); // [2][16][HH]

  int b  = blockIdx.x / 12;
  int ct = blockIdx.x % 12;
  int c0 = ct * 16;
  int tid = threadIdx.x;
  int i0w = (tid >> 5) << 4;       // this wave's 16 rows (N tile)

  int gbase = n * CI + c0;
  v16bf A0 = wmma_load_a(w_wup,   S16, gbase,        0, S16);  // gl rows
  v16bf A1 = wmma_load_a(w_wup,   S16, 768 + gbase,  0, S16);  // gm rows
  v16bf A2 = wmma_load_a(w_wup,   S16, 1536 + gbase, 0, S16);  // gr rows
  v16bf A3 = wmma_load_a(w_lup,   S16, gbase,        0, S16);  // lambda
  v16bf A4 = wmma_load_a(w_uup,   S16, gbase,        0, S16);  // gain
  v16bf A5 = wmma_load_a(w_ddown, S16, gbase,        0, S16);  // skip

  for (int k = tid; k < 16 * HH; k += 256) h[k] = 0.0f;

  const float* xpb = xp  + b * S16 * HW;
  const float* xb  = x1d + b * CI  * HW;
  float*       yb  = y   + b * CI  * HW;
  float wmn = w_m[n];
  int cl = tid >> 4;                // local channel 0..15
  int i0 = (tid & 15) << 3;         // 8 consecutive rows per thread
  int c  = c0 + cl;

  // async prefetch of X column 0 into xcol buffer 0 (per-lane strided gather)
  {
    float* dst = xcol + cl * HH;
#pragma unroll
    for (int r = 0; r < 8; ++r) {
      int i = i0 + r;
      const float* ga = xb + (c << 14) + xoff_dir(n, i, 0);
      unsigned loff = (unsigned)((char*)&dst[i] - smem);
      unsigned long long g64 = (unsigned long long)(uintptr_t)ga;
      asm volatile("global_load_async_to_lds_b32 %0, %1, off"
                   :: "v"(loff), "v"(g64) : "memory");
    }
  }
  asm volatile("s_wait_asynccnt 0" ::: "memory");
  __syncthreads();

  for (int t = 0; t < WW; ++t) {
    // prefetch next column's X while we compute this one
    if (t + 1 < WW) {
      float* dst = xcol + ((t + 1) & 1) * 16 * HH + cl * HH;
#pragma unroll
      for (int r = 0; r < 8; ++r) {
        int i = i0 + r;
        const float* ga = xb + (c << 14) + xoff_dir(n, i, t + 1);
        unsigned loff = (unsigned)((char*)&dst[i] - smem);
        unsigned long long g64 = (unsigned long long)(uintptr_t)ga;
        asm volatile("global_load_async_to_lds_b32 %0, %1, off"
                     :: "v"(loff), "v"(g64) : "memory");
      }
    }

    // gate GEMM for column t: 6 WMMA tiles -> LDS (bf16)
    v16bf Bc = load_b_xpcol(xpb, i0w, t);
    { v8f cc = {}; cc = wmma_bf16(A0, Bc, cc); store_tile_lds_bf16(g + 0 * 16 * HH, i0w, cc); }
    { v8f cc = {}; cc = wmma_bf16(A1, Bc, cc); store_tile_lds_bf16(g + 1 * 16 * HH, i0w, cc); }
    { v8f cc = {}; cc = wmma_bf16(A2, Bc, cc); store_tile_lds_bf16(g + 2 * 16 * HH, i0w, cc); }
    { v8f cc = {}; cc = wmma_bf16(A3, Bc, cc); store_tile_lds_bf16(g + 3 * 16 * HH, i0w, cc); }
    { v8f cc = {}; cc = wmma_bf16(A4, Bc, cc); store_tile_lds_bf16(g + 4 * 16 * HH, i0w, cc); }
    { v8f cc = {}; cc = wmma_bf16(A5, Bc, cc); store_tile_lds_bf16(g + 5 * 16 * HH, i0w, cc); }
    __syncthreads();

    // snapshot previous-column state i0-1 .. i0+8 into registers
    float hp[10];
#pragma unroll
    for (int k = 0; k < 10; ++k) {
      int ii = i0 + k - 1;
      hp[k] = (ii >= 0 && ii < HH) ? h[cl * HH + ii] : 0.0f;
    }
    __syncthreads();

    const float* xc = xcol + (t & 1) * 16 * HH + cl * HH;
#pragma unroll
    for (int r = 0; r < 8; ++r) {
      int i = i0 + r;
      float gl = sigmoidf((float)g[0 * 16 * HH + cl * HH + i]);
      float gm = sigmoidf((float)g[1 * 16 * HH + cl * HH + i]);
      float gr = sigmoidf((float)g[2 * 16 * HH + cl * HH + i]);
      float ssum = (i == 0) ? (gm + gr) : ((i == HH - 1) ? (gl + gm) : (gl + gm + gr));
      ssum = fmaxf(ssum, 1e-7f);
      float sinv = 1.0f / ssum;
      float l = (float)g[3 * 16 * HH + cl * HH + i];
      float u = (float)g[4 * 16 * HH + cl * HH + i];
      float d = (float)g[5 * 16 * HH + cl * HH + i];
      float X = xc[i];
      float hn = l * X + (gl * hp[r] + gm * hp[r + 1] + gr * hp[r + 2]) * sinv;
      h[cl * HH + i] = hn;
      float* yp = yb + (c << 14) + (i << 7) + t;
      float val = wmn * (hn * u + X * d);
      if (accumulate) val += *yp;
      *yp = val;
    }
    // drain this wave's prefetch of column t+1, then rendezvous
    asm volatile("s_wait_asynccnt 0" ::: "memory");
    __syncthreads();
  }
}

// ---------------------------------------------------------------------------
// Kernel E1: channel LayerNorm per pixel (eps 1e-5).
__global__ void k_ln(const float* __restrict__ y, const float* __restrict__ ln_w,
                     const float* __restrict__ ln_b, float* __restrict__ yln) {
  int tid = blockIdx.x * 256 + threadIdx.x;   // 0..32767
  int b = tid >> 14, p = tid & (HW - 1);
  const float* yb = y + b * CI * HW + p;
  float s = 0.0f, sq = 0.0f;
  for (int c = 0; c < CI; ++c) { float v = yb[c << 14]; s += v; sq += v * v; }
  float mu  = s * (1.0f / CI);
  float var = sq * (1.0f / CI) - mu * mu;
  float inv = rsqrtf(var + 1e-5f);
  float* ob = yln + b * CI * HW + p;
  for (int c = 0; c < CI; ++c)
    ob[c << 14] = (yb[c << 14] - mu) * inv * ln_w[c] + ln_b[c];
}

// Kernel E2: per-(b,c) sum of squares (deterministic tree reduction).
__global__ void k_sumsq(const float* __restrict__ yln, float* __restrict__ gsq) {
  int bc = blockIdx.x;                       // 0..383
  const float* p = yln + ((size_t)bc << 14);
  float s = 0.0f;
  for (int idx = threadIdx.x; idx < HW; idx += 256) { float v = p[idx]; s += v * v; }
  __shared__ float red[256];
  red[threadIdx.x] = s;
  __syncthreads();
  for (int off = 128; off; off >>= 1) {
    if (threadIdx.x < (unsigned)off) red[threadIdx.x] += red[threadIdx.x + off];
    __syncthreads();
  }
  if (threadIdx.x == 0) gsq[bc] = red[0];
}

// Kernel E3: GRN scale per (b,c) + beta folded into an out-proj bias.
// y2 = yln*(1 + gamma*Nx) + beta  =>  GEMM input scale + constant bias per out row.
__global__ void k_grn_prep(const float* __restrict__ gsq, const float* __restrict__ gamma,
                           const float* __restrict__ beta, const float* __restrict__ w_out,
                           float* __restrict__ scale, float* __restrict__ biasv) {
  __shared__ float Gx[2 * CI];
  __shared__ float meanb[2];
  int t = threadIdx.x;
  if (t < 2 * CI) Gx[t] = sqrtf(gsq[t]);
  __syncthreads();
  if (t < 2) {
    float s = 0.0f;
    for (int c = 0; c < CI; ++c) s += Gx[t * CI + c];
    meanb[t] = s * (1.0f / CI);
  }
  __syncthreads();
  if (t < 2 * CI) {
    int b = t / CI, c = t % CI;
    float Nx = Gx[t] / (meanb[b] + 1e-6f);
    scale[t] = 1.0f + gamma[c] * Nx;
  }
  if (t < CIN) {
    float s = 0.0f;
    for (int c = 0; c < CI; ++c) s += w_out[t * CI + c] * beta[c];
    biasv[t] = s;
  }
}

// Kernel E4: out = w_out(96x192) @ (yln * scale[b])(192x16384) + bias.
__global__ void k_outproj(const float* __restrict__ yln, const float* __restrict__ w_out,
                          const float* __restrict__ scale, const float* __restrict__ biasv,
                          float* __restrict__ out) {
  int wave = blockIdx.x * (blockDim.x >> 5) + (threadIdx.x >> 5);
  int nt = wave & 1023;
  int rem = wave >> 10;        // 0..11
  int mt = rem % 6;
  int b  = rem / 6;
  const float* Yb = yln + (size_t)b * CI * HW;
  const float* sc = scale + b * CI;
  v8f c = {};
#pragma unroll
  for (int k0 = 0; k0 < CI; k0 += 32) {
    v16bf a  = wmma_load_a(w_out, CI, mt * 16, k0, CI);
    v16bf bb = wmma_load_b_scaled(Yb, HW, k0, nt * 16, CI, sc);
    c = wmma_bf16(a, bb, c);
  }
  wmma_store_bias(out + (size_t)b * CIN * HW, HW, mt * 16, nt * 16, c, biasv);
}

// ---------------------------------------------------------------------------
extern "C" void kernel_launch(void* const* d_in, const int* in_sizes, int n_in,
                              void* d_out, int out_size, void* d_ws, size_t ws_size,
                              hipStream_t stream) {
  const float* x        = (const float*)d_in[0];
  const float* w_in     = (const float*)d_in[1];
  const float* w_dw     = (const float*)d_in[2];
  const float* b_dw     = (const float*)d_in[3];
  const float* w_xdown  = (const float*)d_in[4];
  const float* w_wup    = (const float*)d_in[5];
  const float* w_lup    = (const float*)d_in[6];
  const float* w_uup    = (const float*)d_in[7];
  const float* w_ddown  = (const float*)d_in[8];
  const float* w_m      = (const float*)d_in[9];
  const float* grn_g    = (const float*)d_in[10];
  const float* grn_b    = (const float*)d_in[11];
  const float* ln_w     = (const float*)d_in[12];
  const float* ln_b     = (const float*)d_in[13];
  const float* w_out    = (const float*)d_in[14];
  float* out = (float*)d_out;

  float* ws    = (float*)d_ws;
  float* x1    = ws;                  // 2*192*16384
  float* x1d   = x1  + 2 * CI * HW;   // 2*192*16384
  float* xp    = x1d + 2 * CI * HW;   // 2*16*16384
  float* y     = xp  + 2 * S16 * HW;  // 2*192*16384
  float* gsq   = y   + 2 * CI * HW;   // 384
  float* scale = gsq + 2 * CI;        // 384
  float* biasv = scale + 2 * CI;      // 96
  float* yln   = x1;                  // reuse x1 (dead after dwconv)

  const int scan_lds = 6 * 16 * HH * 2 + 16 * HH * 4 + 2 * 16 * HH * 4; // 49152 B

  // 1) in_proj GEMM: 2*12*1024 tiles, 8 waves/block
  k_inproj<<<3072, 256, 0, stream>>>(x, w_in, x1);
  // 2) depthwise 3x3 + bias
  k_dwconv<<<24576, 256, 0, stream>>>(x1, w_dw, b_dw, x1d);
  // 3) state bottleneck GEMM: 2*1024 tiles
  k_xdown<<<256, 256, 0, stream>>>(x1d, w_xdown, xp);
  // 4) four directional scans, sequential so y-accumulation is deterministic
  for (int n = 0; n < 4; ++n)
    k_scan<<<24, 256, scan_lds, stream>>>(x1d, xp, w_wup, w_lup, w_uup, w_ddown,
                                          w_m, y, n, n > 0 ? 1 : 0);
  // 5) LayerNorm2d
  k_ln<<<128, 256, 0, stream>>>(y, ln_w, ln_b, yln);
  // 6) GRN spatial L2 per (b,c)
  k_sumsq<<<384, 256, 0, stream>>>(yln, gsq);
  // 7) GRN scale + bias fold
  k_grn_prep<<<1, 384, 0, stream>>>(gsq, grn_g, grn_b, w_out, scale, biasv);
  // 8) out_proj GEMM with fused GRN scale/bias: 2*6*1024 tiles
  k_outproj<<<1536, 256, 0, stream>>>(yln, w_out, scale, biasv, out);
}
